// SingleScaleGrid_65541200937488
// MI455X (gfx1250) — matrix operand order repaired and checked by reading, not compiled, not run on hardware
//
#include <hip/hip_runtime.h>
#include <hip/hip_bf16.h>

typedef __attribute__((ext_vector_type(16))) _Float16 v16h;
typedef __attribute__((ext_vector_type(8)))  _Float16 v8h;
typedef __attribute__((ext_vector_type(8)))  float    v8f;

#define RESN   160
#define RES2   (RESN * RESN)
#define RES3   (RESN * RES2)
#define NRAYS  2048
#define NSAMP  558
#define NCHUNK 35            // 35*16 = 560 >= 558
#define WPB    8             // waves (rays) per block
#define NEARF  0.05f
#define FARF   3.5f
#define STEPS  0.5f
#define VOXELF 0.0125f
#define STEPLEN (STEPS * VOXELF)   // 0.00625
#define ACT_SHIFT -4.5951198501345898f

struct WaveLds {
  alignas(16) _Float16 A0[16][32];    // [ind_norm(3) | latent(12) | zero pad]
  alignas(16) _Float16 H1[16][32];    // attention hidden (full 32 written/chunk)
  alignas(16) _Float16 H2[16][32];    // attention hidden (full 32 written/chunk)
  alignas(16) _Float16 H3[16][32];    // 16-wide stages; cols 16..31 zeroed once
  alignas(16) _Float16 H4[16][32];    // 16-wide stages; cols 16..31 zeroed once
  alignas(16) _Float16 Adec[16][64];  // [rgb_latent(15) | emb(39) | zero pad(10)]
  alignas(16) float    logits[16][4];
  alignas(16) float    dens[16];
  alignas(16) float    rgbraw[16][4];
  alignas(16) float    emb[40];
};

// Wave-local LDS handoff: LDS ops from one wave complete in order (ISA: DS
// in-order), so no hardware barrier is needed — only a compiler code-motion /
// memory fence so stores by some lanes aren't moved past loads by others.
__device__ inline void wave_fence() {
  __asm__ volatile("" ::: "memory");
  __builtin_amdgcn_wave_barrier();
  __asm__ volatile("" ::: "memory");
}

__device__ inline v8f wmma16(v16h a, v16h b, v8f c) {
  return __builtin_amdgcn_wmma_f32_16x16x32_f16(false, a, false, b, (short)0, c,
                                                false, false);
}

// A fragment (16x32 f16, M=samples): lanes 0-15 -> K 0-15,
// lanes 16-31 -> K 16-31. Two 16B-aligned ds_load_b128 per lane.
__device__ inline v16h load_a_tile(const _Float16* row, int lane) {
  int klo = (lane < 16) ? 0 : 16;
  v8h lo = *(const v8h*)(row + klo);
  v8h hi = *(const v8h*)(row + klo + 8);
  v16h a;
#pragma unroll
  for (int j = 0; j < 8; ++j) { a[j] = lo[j]; a[8 + j] = hi[j]; }
  return a;
}
__device__ inline v16h load_a32(const _Float16 (*A)[32], int lane) {
  return load_a_tile(&A[lane & 15][0], lane);
}
__device__ inline v16h load_a64(const _Float16 (*A)[64], int k0, int lane) {
  return load_a_tile(&A[lane & 15][k0], lane);
}

// B fragment (32x16 f16, K x N): lane n holds column n, 16 contiguous K values
// (lanes 0-15: K k0..k0+15, lanes 16-31: K k0+16..k0+31). kshift remaps
// A-columns onto weight rows (row = a_col + kshift); out-of-range -> zero.
__device__ inline v16h load_bw(const float* W, int din, int dout, int n0, int k0,
                               int kshift, int lane) {
  int n = (lane & 15) + n0;
  int kb = k0 + ((lane < 16) ? 0 : 16);
  v16h b;
#pragma unroll
  for (int i = 0; i < 16; ++i) {
    int row = kb + i + kshift;
    float w = (row >= 0 && row < din && n < dout) ? W[row * dout + n] : 0.0f;
    b[i] = (_Float16)w;
  }
  return b;
}

// D (16x16 f32): lane n = col, rows rbase..rbase+7. bias + optional relu,
// store f16 row-major into column n (8 ds_store_b16).
__device__ inline void store_h16(_Float16 (*O)[32], v8f d, float bias, int n0,
                                 bool relu_on, int lane) {
  int n = (lane & 15) + n0;
  int rbase = (lane < 16) ? 0 : 8;
#pragma unroll
  for (int r = 0; r < 8; ++r) {
    float v = d[r] + bias;
    if (relu_on) v = fmaxf(v, 0.0f);
    O[rbase + r][n] = (_Float16)v;
  }
}

__global__ __launch_bounds__(256) void nerf_fwd_kernel(
    const float* __restrict__ rays_o, const float* __restrict__ rays_d,
    const float* __restrict__ viewdirs, const float* __restrict__ grid,
    const float* aw1, const float* ab1, const float* aw2, const float* ab2,
    const float* aw3, const float* ab3,
    const float* pw1, const float* pb1, const float* pw2, const float* pb2,
    const float* pw3, const float* pb3,
    const float* dw1, const float* db1, const float* dw2, const float* db2,
    const float* dw3, const float* db3, const float* dw4, const float* db4,
    float* __restrict__ out) {
  __shared__ WaveLds lds[WPB];
  const int wave = threadIdx.x >> 5;
  const int lane = threadIdx.x & 31;
  const int ray  = blockIdx.x * WPB + wave;
  WaveLds& L = lds[wave];

  // ---- per-ray setup (redundant across lanes; cheap) ----
  const float ox = rays_o[ray * 3 + 0], oy = rays_o[ray * 3 + 1], oz = rays_o[ray * 3 + 2];
  const float dx = rays_d[ray * 3 + 0], dy = rays_d[ray * 3 + 1], dz = rays_d[ray * 3 + 2];
  const float v0 = viewdirs[ray * 3 + 0], v1 = viewdirs[ray * 3 + 1], v2 = viewdirs[ray * 3 + 2];
  const float dnorm = sqrtf(dx * dx + dy * dy + dz * dz);

  float tmin, tmax;
  {
    float vecx = (dx == 0.0f) ? 1e-6f : dx;
    float vecy = (dy == 0.0f) ? 1e-6f : dy;
    float vecz = (dz == 0.0f) ? 1e-6f : dz;
    float rax = (1.0f - ox) / vecx, rbx = (-1.0f - ox) / vecx;
    float ray_ = (1.0f - oy) / vecy, rby = (-1.0f - oy) / vecy;
    float raz = (1.0f - oz) / vecz, rbz = (-1.0f - oz) / vecz;
    float mn = fmaxf(fmaxf(fminf(rax, rbx), fminf(ray_, rby)), fminf(raz, rbz));
    float mx = fminf(fminf(fmaxf(rax, rbx), fmaxf(ray_, rby)), fmaxf(raz, rbz));
    tmin = fminf(fmaxf(mn, NEARF), FARF);
    tmax = fminf(fmaxf(mx, NEARF), FARF);
  }
  const bool mask_ray = (tmax <= tmin);

  // positional encoding of viewdirs (per ray, 39 values)
  for (int j = lane; j < 39; j += 32) {
    float e;
    if (j < 3) {
      e = (j == 0) ? v0 : ((j == 1) ? v1 : v2);
    } else {
      int l = (j - 3) / 6, r = (j - 3) % 6;
      float f = (float)(1 << l);
      float x = ((r % 3) == 0) ? v0 : (((r % 3) == 1) ? v1 : v2);
      e = (r < 3) ? sinf(x * f) : cosf(x * f);
    }
    L.emb[j] = e;
  }
  wave_fence();
  // stage emb into decoder A-tile cols 15..53 (constant per ray), zero pad 54..63
  for (int i = lane; i < 16 * 49; i += 32) {
    int m = i / 49, col = 15 + i % 49;
    L.Adec[m][col] = (col < 54) ? (_Float16)L.emb[col - 15] : (_Float16)0.0f;
  }
  // zero K-pad: A0 cols 15..31; H3/H4 cols 16..31 (never rewritten)
  for (int i = lane; i < 16 * 17; i += 32) {
    int m = i / 17, col = 15 + i % 17;
    L.A0[m][col] = (_Float16)0.0f;
  }
  for (int i = lane; i < 16 * 16; i += 32) {
    int m = i / 16, col = 16 + (i & 15);
    L.H3[m][col] = (_Float16)0.0f;
    L.H4[m][col] = (_Float16)0.0f;
  }

  // ---- preload B fragments (weights) into registers ----
  const v16h Ba1a = load_bw(aw1, 15, 32, 0, 0, 0, lane);
  const v16h Ba1b = load_bw(aw1, 15, 32, 16, 0, 0, lane);
  const v16h Ba2a = load_bw(aw2, 32, 32, 0, 0, 0, lane);
  const v16h Ba2b = load_bw(aw2, 32, 32, 16, 0, 0, lane);
  const v16h Ba3  = load_bw(aw3, 32, 4, 0, 0, 0, lane);
  const v16h Bp1  = load_bw(pw1, 12, 16, 0, 0, -3, lane);  // A cols 3..14 = latent
  const v16h Bp2  = load_bw(pw2, 16, 16, 0, 0, 0, lane);
  const v16h Bp3  = load_bw(pw3, 16, 16, 0, 0, 0, lane);
  const v16h Bd1a = load_bw(dw1, 54, 16, 0, 0, 0, lane);
  const v16h Bd1b = load_bw(dw1, 54, 16, 0, 32, 0, lane);
  const v16h Bd2  = load_bw(dw2, 16, 16, 0, 0, 0, lane);
  const v16h Bd3  = load_bw(dw3, 16, 16, 0, 0, 0, lane);
  const v16h Bd4  = load_bw(dw4, 16, 3, 0, 0, 0, lane);
  const int n = lane & 15;
  const float bA1a = ab1[n], bA1b = ab1[16 + n];
  const float bA2a = ab2[n], bA2b = ab2[16 + n];
  const float bA3 = (n < 4) ? ab3[n] : 0.0f;
  const float bP1 = pb1[n], bP2 = pb2[n], bP3 = pb3[n];
  const float bD1 = db1[n], bD2 = db2[n], bD3 = db3[n];
  const float bD4 = (n < 3) ? db4[n] : 0.0f;

  float T = 1.0f;  // running transmittance (cumprod of clamped 1-alpha)
  float accR = 0.0f, accG = 0.0f, accB = 0.0f, accD = 0.0f, accA = 0.0f;

  for (int chunk = 0; chunk < NCHUNK; ++chunk) {
    // ---- sample geometry + trilinear gather, all 32 lanes ----
    // lane l and lane l+16 handle the same sample m = l&15:
    //   lanes 0-15 gather channels 0-5 (+ write ind_norm),
    //   lanes 16-31 gather channels 6-11.
    const int m = lane & 15;
    const int s = chunk * 16 + m;
    const float interpx = tmin + (STEPLEN * (float)s) / dnorm;
    const float px = ox + dx * interpx;
    const float py = oy + dy * interpx;
    const float pz = oz + dz * interpx;
    const bool inb = (!mask_ray) && (s < NSAMP) &&
                     px >= -1.0f && px <= 1.0f && py >= -1.0f && py <= 1.0f &&
                     pz >= -1.0f && pz <= 1.0f;
    {
      float gx = (px + 1.0f) * 0.5f * 159.0f;
      float gy = (py + 1.0f) * 0.5f * 159.0f;
      float gz = (pz + 1.0f) * 0.5f * 159.0f;
      float flx = floorf(gx), fly = floorf(gy), flz = floorf(gz);
      float fx = gx - flx, fy = gy - fly, fz = gz - flz;
      int x0 = (int)fminf(fmaxf(flx, 0.0f), 159.0f);
      int y0 = (int)fminf(fmaxf(fly, 0.0f), 159.0f);
      int z0 = (int)fminf(fmaxf(flz, 0.0f), 159.0f);
      int x1 = (x0 < 159) ? x0 + 1 : 159;
      int y1 = (y0 < 159) ? y0 + 1 : 159;
      int z1 = (z0 < 159) ? z0 + 1 : 159;
      int b00 = x0 * RES2 + y0 * RESN, b01 = x0 * RES2 + y1 * RESN;
      int b10 = x1 * RES2 + y0 * RESN, b11 = x1 * RES2 + y1 * RESN;
      int o000 = b00 + z0, o001 = b00 + z1, o010 = b01 + z0, o011 = b01 + z1;
      int o100 = b10 + z0, o101 = b10 + z1, o110 = b11 + z0, o111 = b11 + z1;
      float w000 = (1 - fx) * (1 - fy) * (1 - fz), w001 = (1 - fx) * (1 - fy) * fz;
      float w010 = (1 - fx) * fy * (1 - fz),       w011 = (1 - fx) * fy * fz;
      float w100 = fx * (1 - fy) * (1 - fz),       w101 = fx * (1 - fy) * fz;
      float w110 = fx * fy * (1 - fz),             w111 = fx * fy * fz;
      if (lane < 16) {  // ind_norm = flip(t01)*2-1 = (pz, py, px)
        L.A0[m][0] = (_Float16)pz;
        L.A0[m][1] = (_Float16)py;
        L.A0[m][2] = (_Float16)px;
      }
      int cbase = (lane < 16) ? 0 : 6;
      const float* gc = grid + (size_t)cbase * RES3;
#pragma unroll
      for (int c = 0; c < 6; ++c) {
        float val = w000 * gc[o000] + w001 * gc[o001] + w010 * gc[o010] +
                    w011 * gc[o011] + w100 * gc[o100] + w101 * gc[o101] +
                    w110 * gc[o110] + w111 * gc[o111];
        L.A0[m][3 + cbase + c] = (_Float16)val;
        gc += RES3;
      }
    }
    wave_fence();

    // ---- attention MLP: 15 -> 32 -> 32 -> 4, softmax ----
    v16h a = load_a32(L.A0, lane);
    v8f c0 = {}, c1 = {};
    c0 = wmma16(a, Ba1a, c0);
    c1 = wmma16(a, Ba1b, c1);
    store_h16(L.H1, c0, bA1a, 0, true, lane);
    store_h16(L.H1, c1, bA1b, 16, true, lane);
    wave_fence();
    a = load_a32(L.H1, lane);
    c0 = (v8f){}; c1 = (v8f){};
    c0 = wmma16(a, Ba2a, c0);
    c1 = wmma16(a, Ba2b, c1);
    store_h16(L.H2, c0, bA2a, 0, true, lane);
    store_h16(L.H2, c1, bA2b, 16, true, lane);
    wave_fence();
    a = load_a32(L.H2, lane);
    c0 = (v8f){};
    c0 = wmma16(a, Ba3, c0);
    {
      int rbase = (lane < 16) ? 0 : 8;
      if (n < 4) {
#pragma unroll
        for (int r = 0; r < 8; ++r) L.logits[rbase + r][n] = c0[r] + bA3;
      }
    }
    wave_fence();
    float att1 = 0.0f;
    if (lane < 16) {
      float l0 = L.logits[m][0], l1 = L.logits[m][1];
      float l2 = L.logits[m][2], l3 = L.logits[m][3];
      float mx = fmaxf(fmaxf(l0, l1), fmaxf(l2, l3));
      float e0 = expf(l0 - mx), e1 = expf(l1 - mx), e2 = expf(l2 - mx), e3 = expf(l3 - mx);
      att1 = e1 / (e0 + e1 + e2 + e3);
    }

    // ---- pos MLP: latent(12) -> 16 -> 16 -> 16 (shifted weights reuse A0) ----
    a = load_a32(L.A0, lane);
    c0 = (v8f){};
    c0 = wmma16(a, Bp1, c0);
    store_h16(L.H3, c0, bP1, 0, true, lane);
    wave_fence();
    a = load_a32(L.H3, lane);
    c0 = (v8f){};
    c0 = wmma16(a, Bp2, c0);
    store_h16(L.H4, c0, bP2, 0, true, lane);
    wave_fence();
    a = load_a32(L.H4, lane);
    c0 = (v8f){};
    c0 = wmma16(a, Bp3, c0);
    {  // col0 = raw density; cols 1..15 = rgb_latent -> decoder A cols 0..14
      int rbase = (lane < 16) ? 0 : 8;
#pragma unroll
      for (int r = 0; r < 8; ++r) {
        float val = c0[r] + bP3;
        int mm = rbase + r;
        if (n == 0) L.dens[mm] = val;
        else        L.Adec[mm][n - 1] = (_Float16)val;
      }
    }
    wave_fence();

    // ---- decoder MLP: 54 -> 16 -> 16 -> 16 -> 3 ----
    {
      v16h a0 = load_a64(L.Adec, 0, lane);
      v16h a1 = load_a64(L.Adec, 32, lane);
      c0 = (v8f){};
      c0 = wmma16(a1, Bd1b, c0);
      c0 = wmma16(a0, Bd1a, c0);
    }
    store_h16(L.H3, c0, bD1, 0, true, lane);
    wave_fence();
    a = load_a32(L.H3, lane);
    c0 = (v8f){};
    c0 = wmma16(a, Bd2, c0);
    store_h16(L.H4, c0, bD2, 0, true, lane);
    wave_fence();
    a = load_a32(L.H4, lane);
    c0 = (v8f){};
    c0 = wmma16(a, Bd3, c0);
    store_h16(L.H3, c0, bD3, 0, true, lane);
    wave_fence();
    a = load_a32(L.H3, lane);
    c0 = (v8f){};
    c0 = wmma16(a, Bd4, c0);
    {
      int rbase = (lane < 16) ? 0 : 8;
      if (n < 3) {
#pragma unroll
        for (int r = 0; r < 8; ++r) L.rgbraw[rbase + r][n] = c0[r] + bD4;
      }
    }
    wave_fence();

    // ---- activation + compositing for these 16 samples ----
    float alpha = 0.0f, cr = 0.0f, cg = 0.0f, cb = 0.0f, depth = 0.0f;
    if (lane < 16) {
      depth = interpx * dnorm;  // ||rays_o - ray_pts||
      if (inb) {
        float density = att1 * L.dens[m];
        float x = density + ACT_SHIFT;
        float sp = fmaxf(x, 0.0f) + log1pf(expf(-fabsf(x)));  // softplus
        alpha = 1.0f - expf(-sp * STEPS);
        cr = 1.0f / (1.0f + expf(-att1 * L.rgbraw[m][0]));
        cg = 1.0f / (1.0f + expf(-att1 * L.rgbraw[m][1]));
        cb = 1.0f / (1.0f + expf(-att1 * L.rgbraw[m][2]));
      }
    }
    float om = fmaxf(1.0f - alpha, 1e-10f);
    float incl = om;  // inclusive prefix product across lanes 0..15 (width 16)
#pragma unroll
    for (int off = 1; off < 16; off <<= 1) {
      float u = __shfl_up(incl, off, 16);
      if ((lane & 15) >= off) incl *= u;
    }
    float excl = __shfl_up(incl, 1, 16);
    if ((lane & 15) == 0) excl = 1.0f;
    float w = T * excl * alpha;
    accR += w * cr; accG += w * cg; accB += w * cb;
    accD += w * depth; accA += w;
    T *= __shfl(incl, 15, 16);
    wave_fence();  // keep next chunk's tile rewrites behind this chunk's reads
  }

  // reduce lanes 0..15 -> lane 0
#pragma unroll
  for (int off = 8; off > 0; off >>= 1) {
    accR += __shfl_down(accR, off, 16);
    accG += __shfl_down(accG, off, 16);
    accB += __shfl_down(accB, off, 16);
    accD += __shfl_down(accD, off, 16);
    accA += __shfl_down(accA, off, 16);
  }
  if (lane == 0) {
    float dm = accD + T * FARF;
    out[ray * 6 + 0] = accR + T * 1.0f;  // BG = 1
    out[ray * 6 + 1] = accG + T * 1.0f;
    out[ray * 6 + 2] = accB + T * 1.0f;
    out[ray * 6 + 3] = dm;
    out[ray * 6 + 4] = 1.0f / dm;
    out[ray * 6 + 5] = accA;
  }
}

extern "C" void kernel_launch(void* const* d_in, const int* in_sizes, int n_in,
                              void* d_out, int out_size, void* d_ws, size_t ws_size,
                              hipStream_t stream) {
  (void)in_sizes; (void)n_in; (void)out_size; (void)d_ws; (void)ws_size;
  const float* rays_o   = (const float*)d_in[0];
  const float* rays_d   = (const float*)d_in[1];
  const float* viewdirs = (const float*)d_in[2];
  const float* grid     = (const float*)d_in[3];
  const float* aw1 = (const float*)d_in[4],  *ab1 = (const float*)d_in[5];
  const float* aw2 = (const float*)d_in[6],  *ab2 = (const float*)d_in[7];
  const float* aw3 = (const float*)d_in[8],  *ab3 = (const float*)d_in[9];
  const float* pw1 = (const float*)d_in[10], *pb1 = (const float*)d_in[11];
  const float* pw2 = (const float*)d_in[12], *pb2 = (const float*)d_in[13];
  const float* pw3 = (const float*)d_in[14], *pb3 = (const float*)d_in[15];
  const float* dw1 = (const float*)d_in[16], *db1 = (const float*)d_in[17];
  const float* dw2 = (const float*)d_in[18], *db2 = (const float*)d_in[19];
  const float* dw3 = (const float*)d_in[20], *db3 = (const float*)d_in[21];
  const float* dw4 = (const float*)d_in[22], *db4 = (const float*)d_in[23];
  float* out = (float*)d_out;

  dim3 grid_dim(NRAYS / WPB);  // 256 blocks, one wave per ray
  dim3 block_dim(32 * WPB);    // 256 threads = 8 waves
  nerf_fwd_kernel<<<grid_dim, block_dim, 0, stream>>>(
      rays_o, rays_d, viewdirs, grid,
      aw1, ab1, aw2, ab2, aw3, ab3,
      pw1, pb1, pw2, pb2, pw3, pb3,
      dw1, db1, dw2, db2, dw3, db3, dw4, db4, out);
}
